// GatedDQN_88716844467015
// MI455X (gfx1250) — matrix operand order repaired and checked by reading, not compiled, not run on hardware
//
#include <hip/hip_runtime.h>

// ---------------------------------------------------------------------------
// GatedDQN fused inference for gfx1250 (MI455X, wave32, WMMA).
// out = concat(q_values[B,4], gate_values[B,32]) fp32.
// ---------------------------------------------------------------------------

typedef __attribute__((ext_vector_type(16))) _Float16 v16h;
typedef __attribute__((ext_vector_type(8)))  _Float16 v8h;
typedef __attribute__((ext_vector_type(8)))  float    v8f;
typedef __attribute__((ext_vector_type(4)))  float    v4f;

#define BATCH   65536
#define SACT_LD 132   // fp32 activation row stride (pad vs 128 -> no bank conflicts)
#define SA_LD   136   // fp16 A-staging row stride (272B -> 4-bank rotation per row)

// ---------------- WMMA helpers ---------------------------------------------

static __device__ __forceinline__ v8f wmma16(v16h a, v16h b, v8f c) {
  // D = A(16x32 f16) * B(32x16 f16) + C(16x16 f32)
  return __builtin_amdgcn_wmma_f32_16x16x32_f16(
      /*neg_a=*/false, a, /*neg_b=*/false, b,
      /*c_mod=*/(short)0, c, /*reuse_a=*/false, /*reuse_b=*/false);
}

// A fragment (16x32, fp16) from a global fp32 row-major tile.
// ISA layout: lane l holds row m=l&15; elements 0..7 -> K = g*8+e,
// elements 8..15 -> K = 16+g*8+(e-8), g = l>>4.  Two contiguous 8-float runs.
static __device__ __forceinline__ v16h load_a_f32(const float* __restrict__ A,
                                                  int ldk, int m, int kb, int g) {
  const float* p = A + (size_t)m * ldk + kb + g * 8;
  v4f f0 = __builtin_nontemporal_load((const v4f*)(p + 0));
  v4f f1 = __builtin_nontemporal_load((const v4f*)(p + 4));
  v4f f2 = __builtin_nontemporal_load((const v4f*)(p + 16));
  v4f f3 = __builtin_nontemporal_load((const v4f*)(p + 20));
  v16h a;
  a[0]=(_Float16)f0[0];  a[1]=(_Float16)f0[1];  a[2]=(_Float16)f0[2];  a[3]=(_Float16)f0[3];
  a[4]=(_Float16)f1[0];  a[5]=(_Float16)f1[1];  a[6]=(_Float16)f1[2];  a[7]=(_Float16)f1[3];
  a[8]=(_Float16)f2[0];  a[9]=(_Float16)f2[1];  a[10]=(_Float16)f2[2]; a[11]=(_Float16)f2[3];
  a[12]=(_Float16)f3[0]; a[13]=(_Float16)f3[1]; a[14]=(_Float16)f3[2]; a[15]=(_Float16)f3[3];
  return a;
}

// A fragment from LDS fp16 staging buffer (same layout, two ds_load_b128).
static __device__ __forceinline__ v16h load_a_lds(const _Float16* A, int ld,
                                                  int m, int kb, int g) {
  const _Float16* p = A + m * ld + kb + g * 8;
  v8h lo = *(const v8h*)(p);
  v8h hi = *(const v8h*)(p + 16);
  return __builtin_shufflevector(lo, hi, 0,1,2,3,4,5,6,7,8,9,10,11,12,13,14,15);
}

// B fragment from pre-packed fp16 weights: one contiguous 32B/lane load.
static __device__ __forceinline__ v16h load_b(const _Float16* __restrict__ Bp,
                                              int n_tiles, int kt, int nt, int lane) {
  return *(const v16h*)(Bp + (((kt * n_tiles + nt) * 32 + lane) << 4));
}

// ---------------- weight pre-pack kernel -----------------------------------
// Packs W[N,K] fp32 row-major into per-fragment fp16 order matching the ISA
// 16-bit B (KxN) VGPR layout: frag = kt*n_tiles+nt; lane n=lane&15,
// element e -> K = kt*32 + (lane>>4)*8 + (e&7) + (e>>3)*16.
__global__ void gdqn_pack_w(const float* __restrict__ W, _Float16* __restrict__ dst,
                            int N, int K, int total) {
  int tid = blockIdx.x * blockDim.x + threadIdx.x;
  if (tid >= total) return;
  int f = tid >> 9, r = tid & 511;
  int lane = r >> 4, e = r & 15;
  int n_tiles = N >> 4;
  int nt = f % n_tiles, kt = f / n_tiles;
  int n = nt * 16 + (lane & 15);
  int k = kt * 32 + (lane >> 4) * 8 + (e & 7) + (e >> 3) * 16;
  dst[tid] = (_Float16)W[(size_t)n * K + k];
}

// ---------------- LayerNorm row helper --------------------------------------
template <int F>
static __device__ __forceinline__ void ln_row(const float* row, _Float16* orow,
                                              const float* gg, const float* bb) {
  float s = 0.f, s2 = 0.f;
#pragma unroll 8
  for (int c = 0; c < F; ++c) { float v = row[c]; s += v; s2 += v * v; }
  float mean = s * (1.f / F);
  float var  = s2 * (1.f / F) - mean * mean;
  float ri   = rsqrtf(var + 1e-5f);
#pragma unroll 8
  for (int c = 0; c < F; ++c)
    orow[c] = (_Float16)((row[c] - mean) * ri * gg[c] + bb[c]);
}

// ---------------- main fused kernel -----------------------------------------
__global__ __launch_bounds__(128)
void gdqn_main(const float* __restrict__ x, const float* __restrict__ hist,
               const _Float16* __restrict__ g1p,  const float* __restrict__ g1_b,
               const _Float16* __restrict__ g2p,  const float* __restrict__ g2_b,
               const _Float16* __restrict__ basep,const float* __restrict__ base_b,
               const float* __restrict__ ln1_g,   const float* __restrict__ ln1_b,
               const _Float16* __restrict__ n1p,  const float* __restrict__ n1_b,
               const float* __restrict__ ln2_g,   const float* __restrict__ ln2_b,
               const _Float16* __restrict__ mp,   const float* __restrict__ m_b,
               const float* __restrict__ v1_w, const float* __restrict__ v1_b,
               const float* __restrict__ v2_w, const float* __restrict__ v2_b,
               const float* __restrict__ a1_w, const float* __restrict__ a1_b,
               const float* __restrict__ a2_w, const float* __restrict__ a2_b,
               float* __restrict__ out_q, float* __restrict__ out_gate) {
  __shared__ float    sact[4][16][SACT_LD];   // fp32 activations (per wave)
  __shared__ _Float16 sa[4][16][SA_LD];       // fp16 A-staging   (per wave)
  __shared__ float    smod[4][16];
  __shared__ float shv1[512], sha1[512], sha2[64], shv2[16];
  __shared__ float shv1b[16], sha1b[16], sha2b[4], shv2b[1];
  __shared__ float sln1g[128], sln1b[128], sln2g[64], sln2b[64];

  const int tid  = threadIdx.x;
  const int wave = tid >> 5;
  const int lane = tid & 31;
  const int m15  = lane & 15;
  const int g    = lane >> 4;
  const int row0 = (blockIdx.x * 4 + wave) * 16;

  // stage small parameters into LDS once
  for (int i = tid; i < 512; i += 128) { shv1[i] = v1_w[i]; sha1[i] = a1_w[i]; }
  sln1g[tid] = ln1_g[tid]; sln1b[tid] = ln1_b[tid];
  if (tid < 64) { sln2g[tid] = ln2_g[tid]; sln2b[tid] = ln2_b[tid]; sha2[tid] = a2_w[tid]; }
  if (tid < 16) { shv1b[tid] = v1_b[tid]; sha1b[tid] = a1_b[tid]; shv2[tid] = v2_w[tid]; }
  if (tid < 4)  sha2b[tid] = a2_b[tid];
  if (tid == 0) shv2b[0] = v2_b[0];
  __syncthreads();

  float*    sactw = &sact[wave][0][0];
  _Float16* saw   = &sa[wave][0][0];
  float*    smodw = &smod[wave][0];
  const v8f vzero = {0.f, 0.f, 0.f, 0.f, 0.f, 0.f, 0.f, 0.f};

  // -------- gate layer 1: relu(history[16,2048] @ g1^T[2048,64] + b) --------
  {
    v8f acc[4];
#pragma unroll
    for (int nt = 0; nt < 4; ++nt) acc[nt] = vzero;
    const float* A = hist + (size_t)row0 * 2048;
    for (int kt = 0; kt < 64; ++kt) {
      v16h a = load_a_f32(A, 2048, m15, kt * 32, g);
#pragma unroll
      for (int nt = 0; nt < 4; ++nt)
        acc[nt] = wmma16(a, load_b(g1p, 4, kt, nt, lane), acc[nt]);
    }
#pragma unroll
    for (int nt = 0; nt < 4; ++nt) {
      float bn = g1_b[nt * 16 + m15];
#pragma unroll
      for (int r = 0; r < 8; ++r) {
        float v = acc[nt][r] + bn;
        v = v > 0.f ? v : 0.f;
        saw[(8 * g + r) * SA_LD + nt * 16 + m15] = (_Float16)v;   // fp16 A stage
      }
    }
  }
  __syncthreads();

  // -------- gate layer 2: sigmoid(h[16,64] @ g2^T[64,32] + b) ---------------
  {
    v8f acc[2];
#pragma unroll
    for (int nt = 0; nt < 2; ++nt) acc[nt] = vzero;
#pragma unroll
    for (int kt = 0; kt < 2; ++kt) {
      v16h a = load_a_lds(saw, SA_LD, m15, kt * 32, g);
#pragma unroll
      for (int nt = 0; nt < 2; ++nt)
        acc[nt] = wmma16(a, load_b(g2p, 2, kt, nt, lane), acc[nt]);
    }
#pragma unroll
    for (int nt = 0; nt < 2; ++nt) {
      float bn = g2_b[nt * 16 + m15];
#pragma unroll
      for (int r = 0; r < 8; ++r) {
        int mr = 8 * g + r;
        float v = acc[nt][r] + bn;
        float s = 1.f / (1.f + __expf(-v));
        sactw[mr * SACT_LD + nt * 16 + m15] = s;
        out_gate[(size_t)(row0 + mr) * 32 + nt * 16 + m15] = s;   // output #2
      }
    }
  }
  __syncthreads();
  if (lane < 16) {
    float s = 0.f;
#pragma unroll
    for (int c = 0; c < 32; ++c) s += sactw[lane * SACT_LD + c];
    smodw[lane] = 0.7f + 0.3f * (s * (1.f / 32.f));
  }
  __syncthreads();

  // -------- base: relu((x[16,512] @ base^T[512,128]) * mod + b), LN ---------
  {
    v8f acc[8];
#pragma unroll
    for (int nt = 0; nt < 8; ++nt) acc[nt] = vzero;
    const float* A = x + (size_t)row0 * 512;
    for (int kt = 0; kt < 16; ++kt) {
      v16h a = load_a_f32(A, 512, m15, kt * 32, g);
#pragma unroll
      for (int nt = 0; nt < 8; ++nt)
        acc[nt] = wmma16(a, load_b(basep, 8, kt, nt, lane), acc[nt]);
    }
#pragma unroll
    for (int nt = 0; nt < 8; ++nt) {
      float bn = base_b[nt * 16 + m15];
#pragma unroll
      for (int r = 0; r < 8; ++r) {
        int mr = 8 * g + r;
        float v = acc[nt][r] * smodw[mr] + bn;
        v = v > 0.f ? v : 0.f;
        sactw[mr * SACT_LD + nt * 16 + m15] = v;
      }
    }
  }
  __syncthreads();
  if (lane < 16)
    ln_row<128>(sactw + lane * SACT_LD, saw + lane * SA_LD, sln1g, sln1b);
  __syncthreads();

  // -------- n1: relu(xg[16,128] @ n1^T[128,64] + b), LN ---------------------
  {
    v8f acc[4];
#pragma unroll
    for (int nt = 0; nt < 4; ++nt) acc[nt] = vzero;
#pragma unroll
    for (int kt = 0; kt < 4; ++kt) {
      v16h a = load_a_lds(saw, SA_LD, m15, kt * 32, g);
#pragma unroll
      for (int nt = 0; nt < 4; ++nt)
        acc[nt] = wmma16(a, load_b(n1p, 4, kt, nt, lane), acc[nt]);
    }
#pragma unroll
    for (int nt = 0; nt < 4; ++nt) {
      float bn = n1_b[nt * 16 + m15];
#pragma unroll
      for (int r = 0; r < 8; ++r) {
        float v = acc[nt][r] + bn;
        v = v > 0.f ? v : 0.f;
        sactw[(8 * g + r) * SACT_LD + nt * 16 + m15] = v;
      }
    }
  }
  __syncthreads();
  if (lane < 16)
    ln_row<64>(sactw + lane * SACT_LD, saw + lane * SA_LD, sln2g, sln2b);
  __syncthreads();

  // -------- m: features = relu(f[16,64] @ m^T[64,32] + b) -------------------
  {
    v8f acc[2];
#pragma unroll
    for (int nt = 0; nt < 2; ++nt) acc[nt] = vzero;
#pragma unroll
    for (int kt = 0; kt < 2; ++kt) {
      v16h a = load_a_lds(saw, SA_LD, m15, kt * 32, g);
#pragma unroll
      for (int nt = 0; nt < 2; ++nt)
        acc[nt] = wmma16(a, load_b(mp, 2, kt, nt, lane), acc[nt]);
    }
#pragma unroll
    for (int nt = 0; nt < 2; ++nt) {
      float bn = m_b[nt * 16 + m15];
#pragma unroll
      for (int r = 0; r < 8; ++r) {
        float v = acc[nt][r] + bn;
        sactw[(8 * g + r) * SACT_LD + nt * 16 + m15] = v > 0.f ? v : 0.f;
      }
    }
  }
  __syncthreads();

  // -------- dueling heads (tiny: scalar, one lane per row) ------------------
  if (lane < 16) {
    const float* feat = sactw + lane * SACT_LD;        // cols 0..31
    float* hid = sactw + lane * SACT_LD + 48;          // scratch cols 48..63
    // value stream 32->16->1
    for (int j = 0; j < 16; ++j) {
      float s = shv1b[j];
#pragma unroll 8
      for (int k = 0; k < 32; ++k) s += feat[k] * shv1[j * 32 + k];
      hid[j] = s > 0.f ? s : 0.f;
    }
    float val = shv2b[0];
#pragma unroll
    for (int j = 0; j < 16; ++j) val += hid[j] * shv2[j];
    // advantage stream 32->16->4
    for (int j = 0; j < 16; ++j) {
      float s = sha1b[j];
#pragma unroll 8
      for (int k = 0; k < 32; ++k) s += feat[k] * sha1[j * 32 + k];
      hid[j] = s > 0.f ? s : 0.f;
    }
    float adv[4]; float am = 0.f;
#pragma unroll
    for (int c = 0; c < 4; ++c) {
      float s = sha2b[c];
#pragma unroll
      for (int j = 0; j < 16; ++j) s += hid[j] * sha2[c * 16 + j];
      adv[c] = s; am += s;
    }
    am *= 0.25f;
    float* oq = out_q + (size_t)(row0 + lane) * 4;
#pragma unroll
    for (int c = 0; c < 4; ++c) oq[c] = val + adv[c] - am;
  }
}

// ---------------- launcher ---------------------------------------------------
extern "C" void kernel_launch(void* const* d_in, const int* in_sizes, int n_in,
                              void* d_out, int out_size, void* d_ws, size_t ws_size,
                              hipStream_t stream) {
  const float* x      = (const float*)d_in[0];
  const float* hist   = (const float*)d_in[1];
  const float* g1_w   = (const float*)d_in[2];
  const float* g1_b   = (const float*)d_in[3];
  const float* g2_w   = (const float*)d_in[4];
  const float* g2_b   = (const float*)d_in[5];
  const float* base_w = (const float*)d_in[6];
  const float* base_b = (const float*)d_in[7];
  const float* ln1_g  = (const float*)d_in[8];
  const float* ln1_b  = (const float*)d_in[9];
  const float* n1_w   = (const float*)d_in[10];
  const float* n1_b   = (const float*)d_in[11];
  const float* ln2_g  = (const float*)d_in[12];
  const float* ln2_b  = (const float*)d_in[13];
  const float* m_w    = (const float*)d_in[14];
  const float* m_b    = (const float*)d_in[15];
  const float* v1_w   = (const float*)d_in[16];
  const float* v1_b   = (const float*)d_in[17];
  const float* v2_w   = (const float*)d_in[18];
  const float* v2_b   = (const float*)d_in[19];
  const float* a1_w   = (const float*)d_in[20];
  const float* a1_b   = (const float*)d_in[21];
  const float* a2_w   = (const float*)d_in[22];
  const float* a2_b   = (const float*)d_in[23];

  float* out = (float*)d_out;

  // fp16 packed-weight workspace layout (halfs)
  _Float16* ws    = (_Float16*)d_ws;
  _Float16* g1p   = ws;                       // 64*2048  = 131072
  _Float16* g2p   = g1p + 64 * 2048;          // 32*64    =   2048
  _Float16* basep = g2p + 32 * 64;            // 128*512  =  65536
  _Float16* n1p   = basep + 128 * 512;        // 64*128   =   8192
  _Float16* mpk   = n1p + 64 * 128;           // 32*64    =   2048

  struct PW { const float* w; _Float16* d; int N, K; };
  const PW packs[5] = {
    { g1_w,   g1p,   64, 2048 },
    { g2_w,   g2p,   32,   64 },
    { base_w, basep,128,  512 },
    { n1_w,   n1p,   64,  128 },
    { m_w,    mpk,   32,   64 },
  };
  for (int i = 0; i < 5; ++i) {
    int total = packs[i].N * packs[i].K;
    gdqn_pack_w<<<(total + 255) / 256, 256, 0, stream>>>(
        packs[i].w, packs[i].d, packs[i].N, packs[i].K, total);
  }

  gdqn_main<<<BATCH / 64, 128, 0, stream>>>(
      x, hist, g1p, g1_b, g2p, g2_b, basep, base_b, ln1_g, ln1_b,
      n1p, n1_b, ln2_g, ln2_b, mpk, m_b,
      v1_w, v1_b, v2_w, v2_b, a1_w, a1_b, a2_w, a2_b,
      out /*q [B,4]*/, out + (size_t)BATCH * 4 /*gate [B,32]*/);
}
